// CausalSelfAttentionWithBias_18751827214687
// MI455X (gfx1250) — matrix operand order repaired and checked by reading, not compiled, not run on hardware
//
#include <hip/hip_runtime.h>

typedef __bf16 bf16;
typedef __attribute__((ext_vector_type(16))) __bf16 v16bf;
typedef __attribute__((ext_vector_type(8)))  __bf16 v8bf;
typedef __attribute__((ext_vector_type(8)))  float   v8f;
typedef __attribute__((ext_vector_type(4)))  int     v4i;

#define B_ 4
#define T_ 2048
#define C_ 1024
#define H_ 16
#define D_ 64

#define BM 128
#define BN 128
#define BK 32

#if __has_builtin(__builtin_amdgcn_global_load_async_to_lds_b128)
#define HAVE_ASYNC_LDS 1
#else
#define HAVE_ASYNC_LDS 0
#endif

#define AS1 __attribute__((address_space(1)))
#define AS3 __attribute__((address_space(3)))

// 16-byte global -> LDS copy via CDNA5 async-to-LDS (ASYNCcnt, no VGPR bounce).
__device__ __forceinline__ void g2l_16B(const bf16* g, bf16* l) {
#if HAVE_ASYNC_LDS
  __builtin_amdgcn_global_load_async_to_lds_b128(
      (AS1 v4i*)g, (AS3 v4i*)l, 0, 0);
#else
  *(v8bf*)l = *(const v8bf*)g;
#endif
}

__device__ __forceinline__ void wait_async_lds() {
#if HAVE_ASYNC_LDS
#if __has_builtin(__builtin_amdgcn_s_wait_asynccnt)
  __builtin_amdgcn_s_wait_asynccnt(0);
#else
  asm volatile("s_wait_asynccnt 0x0" ::: "memory");
#endif
#endif
}

// 16-lane xor-butterfly reductions in pure DPP16 VALU (masks 1,2,7,15 span
// GF(2)^4, so every lane of each 16-group ends with the full reduction).
#if __has_builtin(__builtin_amdgcn_update_dpp)
template <int CTRL>
__device__ __forceinline__ float dpp_perm(float x) {
  int xi = __builtin_bit_cast(int, x);
  int yi = __builtin_amdgcn_update_dpp(xi, xi, CTRL, 0xf, 0xf, true);
  return __builtin_bit_cast(float, yi);
}
__device__ __forceinline__ float red16_max(float v) {
  v = fmaxf(v, dpp_perm<0xB1>(v));    // quad_perm [1,0,3,2]  (xor 1)
  v = fmaxf(v, dpp_perm<0x4E>(v));    // quad_perm [2,3,0,1]  (xor 2)
  v = fmaxf(v, dpp_perm<0x141>(v));   // row_half_mirror      (xor 7)
  v = fmaxf(v, dpp_perm<0x140>(v));   // row_mirror           (xor 15)
  return v;
}
__device__ __forceinline__ float red16_sum(float v) {
  v += dpp_perm<0xB1>(v);
  v += dpp_perm<0x4E>(v);
  v += dpp_perm<0x141>(v);
  v += dpp_perm<0x140>(v);
  return v;
}
#else
__device__ __forceinline__ float red16_max(float v) {
  v = fmaxf(v, __shfl_xor(v, 1)); v = fmaxf(v, __shfl_xor(v, 2));
  v = fmaxf(v, __shfl_xor(v, 4)); v = fmaxf(v, __shfl_xor(v, 8));
  return v;
}
__device__ __forceinline__ float red16_sum(float v) {
  v += __shfl_xor(v, 1); v += __shfl_xor(v, 2);
  v += __shfl_xor(v, 4); v += __shfl_xor(v, 8);
  return v;
}
#endif

__device__ __forceinline__ v16bf mk16(v8bf lo, v8bf hi) {
  union { v16bf v; v8bf h[2]; } u;
  u.h[0] = lo; u.h[1] = hi;
  return u.v;
}

__device__ __forceinline__ v8f wmma_bf16(v16bf a, v16bf b, v8f c) {
  return __builtin_amdgcn_wmma_f32_16x16x32_bf16(false, a, false, b, (short)0, c, false, false);
}

// ---------------------------------------------------------------------------
// Kernel 1: one-time fp32 -> bf16 of x (copy) and weights (transposed [N][K]).
// ---------------------------------------------------------------------------
__global__ __launch_bounds__(256) void convert_inputs(
    const float* __restrict__ X,  const float* __restrict__ Wa,
    const float* __restrict__ Wp, bf16* __restrict__ Xb,
    bf16* __restrict__ WaT, bf16* __restrict__ WpT) {
  int stride = gridDim.x * blockDim.x;
  int i0 = blockIdx.x * blockDim.x + threadIdx.x;
  const int nX = B_ * T_ * C_;
  for (int i = i0; i < nX; i += stride) Xb[i] = (bf16)X[i];
  const int Na = 3 * C_;
  for (int i = i0; i < C_ * Na; i += stride) {
    int k = i / Na, n = i % Na;
    WaT[(size_t)n * C_ + k] = (bf16)Wa[i];
  }
  for (int i = i0; i < C_ * C_; i += stride) {
    int k = i / C_, n = i % C_;
    WpT[(size_t)n * C_ + k] = (bf16)Wp[i];
  }
}

// ---------------------------------------------------------------------------
// Kernel 2: bf16 WMMA GEMM, 128x128x32 tiles, 8 waves (2x4), 64x32 per wave.
// Double-buffered LDS: async copies for tile k+1 are issued BEFORE computing
// tile k and waited only at the end of the iteration (latency hidden).
// MODE 0: epilogue adds b_attn + head bias; writes Q (pre-scaled by 1/8),
//         K as [B,H,T,D] and V as [B,H,D,T]. MODE 1: adds b_proj, fp32 out.
// ---------------------------------------------------------------------------
template <int MODE>
__global__ __launch_bounds__(256) void gemm_wmma(
    const bf16* __restrict__ Ab,  const bf16* __restrict__ BwT,
    const float* __restrict__ bias,
    const float* __restrict__ bQ, const float* __restrict__ bK,
    const float* __restrict__ bV,
    bf16* __restrict__ Qo, bf16* __restrict__ Ko, bf16* __restrict__ Vo,
    float* __restrict__ Co, int M, int N, int K) {
  __shared__ __align__(16) bf16 As[2][BM][BK];   // 16 KB
  __shared__ __align__(16) bf16 Bs[2][BN][BK];   // 16 KB

  const int tid  = threadIdx.x;
  const int wv   = tid >> 5;
  const int lane = tid & 31;
  const int ln16 = lane & 15;
  const int hl   = lane >> 4;

  const int tileM = blockIdx.y * BM;
  const int tileN = blockIdx.x * BN;
  const int wm = (wv >> 2) * 64;
  const int wn = (wv & 3) * 32;

  const int r = tid >> 1;            // 0..127 stage row
  const int c = (tid & 1) * 16;      // 0 / 16 stage col

  auto stage = [&](int buf, int k0) {
    const bf16* sa = Ab  + (size_t)(tileM + r) * K + k0 + c;
    const bf16* sb = BwT + (size_t)(tileN + r) * K + k0 + c;
    g2l_16B(sa,     &As[buf][r][c]);
    g2l_16B(sa + 8, &As[buf][r][c + 8]);
    g2l_16B(sb,     &Bs[buf][r][c]);
    g2l_16B(sb + 8, &Bs[buf][r][c + 8]);
  };

  v8f acc[4][2];
#pragma unroll
  for (int i = 0; i < 4; ++i)
#pragma unroll
    for (int j = 0; j < 2; ++j) acc[i][j] = (v8f){};

  stage(0, 0);
  wait_async_lds();
  __syncthreads();

  int cur = 0;
  for (int k0 = 0; k0 < K; k0 += BK) {
    if (k0 + BK < K) stage(cur ^ 1, k0 + BK);   // async prefetch, no wait

    v16bf af[4], bfr[2];
#pragma unroll
    for (int i = 0; i < 4; ++i) {
      int m = wm + i * 16 + ln16;
      af[i] = mk16(*(const v8bf*)&As[cur][m][hl * 16],
                   *(const v8bf*)&As[cur][m][hl * 16 + 8]);
    }
#pragma unroll
    for (int j = 0; j < 2; ++j) {
      int n = wn + j * 16 + ln16;
      bfr[j] = mk16(*(const v8bf*)&Bs[cur][n][hl * 16],
                    *(const v8bf*)&Bs[cur][n][hl * 16 + 8]);
    }
#pragma unroll
    for (int i = 0; i < 4; ++i)
#pragma unroll
      for (int j = 0; j < 2; ++j)
        acc[i][j] = wmma_bf16(af[i], bfr[j], acc[i][j]);

    wait_async_lds();      // next buffer written
    __syncthreads();       // and everyone done reading current buffer
    cur ^= 1;
  }

  // ---- epilogue: C layout element rr -> (M = rr + 8*hl, N = ln16) ----
#pragma unroll
  for (int j = 0; j < 2; ++j) {
    int gn = tileN + wn + j * 16 + ln16;
    if (MODE == 0) {
      int which = gn / C_;
      int cc    = gn % C_;
      int h     = cc / D_;
      int d     = cc % D_;
      float bb  = bias[gn] +
                  (which == 0 ? bQ : which == 1 ? bK : bV)[h * D_ + d];
      if (which == 2) {
        // V stored transposed [B,H,D,T]: consecutive rr -> consecutive t
#pragma unroll
        for (int i = 0; i < 4; ++i)
#pragma unroll
          for (int rr = 0; rr < 8; ++rr) {
            int gm = tileM + wm + i * 16 + 8 * hl + rr;
            int b = gm >> 11, t = gm & (T_ - 1);
            Vo[(((size_t)b * H_ + h) * D_ + d) * T_ + t] =
                (bf16)(acc[i][j][rr] + bb);
          }
      } else {
        bf16* dst = (which == 0 ? Qo : Ko);    // [B,H,T,D]
        const float sc = (which == 0) ? 0.125f : 1.0f;  // fold 1/sqrt(D) into Q
#pragma unroll
        for (int i = 0; i < 4; ++i)
#pragma unroll
          for (int rr = 0; rr < 8; ++rr) {
            int gm = tileM + wm + i * 16 + 8 * hl + rr;
            int b = gm >> 11, t = gm & (T_ - 1);
            dst[(((size_t)b * H_ + h) * T_ + t) * D_ + d] =
                (bf16)((acc[i][j][rr] + bb) * sc);
          }
      }
    } else {
      float bb = bias[gn];
#pragma unroll
      for (int i = 0; i < 4; ++i)
#pragma unroll
        for (int rr = 0; rr < 8; ++rr) {
          int gm = tileM + wm + i * 16 + 8 * hl + rr;
          Co[(size_t)gm * N + gn] = acc[i][j][rr] + bb;
        }
    }
  }
}

// ---------------------------------------------------------------------------
// Kernel 3: flash attention, (b,h, 128 queries) per WG; 8 waves x 16 q.
// Double-buffered async staging of K [128][64] and V^T [64][128] blocks.
// Per 32-key chunk: S = Q K^T (2 WMMAs over D=64; Q pre-scaled), causal mask,
// online softmax (DPP butterfly reductions), P via per-wave LDS scratch,
// O += P V (4 WMMAs, 32-key contraction).
// ---------------------------------------------------------------------------
__global__ __launch_bounds__(256) void flash_attn(
    const bf16* __restrict__ Qb, const bf16* __restrict__ Kb,
    const bf16* __restrict__ VtG, bf16* __restrict__ Yb) {
  __shared__ __align__(16) bf16 Ks[2][128][D_];   // 32 KB [key][d]
  __shared__ __align__(16) bf16 Vt[2][D_][128];   // 32 KB [d][key]
  __shared__ __align__(16) bf16 Ps[8][16][32];    //  8 KB per-wave P scratch

  const int tid  = threadIdx.x;
  const int wv   = tid >> 5;
  const int lane = tid & 31;
  const int ln16 = lane & 15;
  const int hl   = lane >> 4;

  const int bh   = blockIdx.y;            // b*H + h
  const int bIdx = bh / H_;
  const int h    = bh % H_;
  const int qblk = blockIdx.x;
  const int q0   = qblk * 128 + wv * 16;

  const int kr  = tid >> 1;               // 0..127 key row
  const int kc0 = (tid & 1) * 32;         // d col
  const int vd  = tid >> 2;               // 0..63  d row
  const int vc0 = (tid & 3) * 32;         // key col

  auto stage = [&](int buf, int kb0) {
    const bf16* ksrc = Kb  + ((size_t)bh * T_ + kb0 + kr) * D_ + kc0;
    const bf16* vsrc = VtG + ((size_t)bh * D_ + vd) * T_ + kb0 + vc0;
#pragma unroll
    for (int j = 0; j < 4; ++j) {
      g2l_16B(ksrc + j * 8, &Ks[buf][kr][kc0 + j * 8]);
      g2l_16B(vsrc + j * 8, &Vt[buf][vd][vc0 + j * 8]);
    }
  };

  // ---- load Q fragments once (A layout, D=64 -> 2 fragments of K=32) ----
  const bf16* qrow = Qb + ((size_t)bh * T_ + q0 + ln16) * D_;
  v16bf aq[2];
#pragma unroll
  for (int f = 0; f < 2; ++f)
    aq[f] = mk16(*(const v8bf*)(qrow + f * 32 + hl * 16),
                 *(const v8bf*)(qrow + f * 32 + hl * 16 + 8));

  v8f o_acc[4];
#pragma unroll
  for (int i = 0; i < 4; ++i) o_acc[i] = (v8f){};
  float m_run[8], l_run[8];
#pragma unroll
  for (int rr = 0; rr < 8; ++rr) { m_run[rr] = -1e30f; l_run[rr] = 0.0f; }

  const int nblocks = qblk + 1;           // causal: key blocks 0..qblk
  stage(0, 0);
  wait_async_lds();
  __syncthreads();

  int cur = 0;
  for (int kb = 0; kb < nblocks; ++kb) {
    const int kb0 = kb * 128;
    if (kb + 1 < nblocks) stage(cur ^ 1, kb0 + 128);  // async prefetch

    for (int cch = 0; cch < 4; ++cch) {
      const int kbase = kb0 + cch * 32;
      if (kbase > q0 + 15) break;         // fully masked for this wave

      // ---- S = Q K^T for two 16-key subtiles (Q pre-scaled by 1/8) ----
      v8f s[2];
#pragma unroll
      for (int t = 0; t < 2; ++t) {
        int kloc = cch * 32 + t * 16 + ln16;
        v16bf bk0 = mk16(*(const v8bf*)&Ks[cur][kloc][hl * 16],
                         *(const v8bf*)&Ks[cur][kloc][hl * 16 + 8]);
        v16bf bk1 = mk16(*(const v8bf*)&Ks[cur][kloc][32 + hl * 16],
                         *(const v8bf*)&Ks[cur][kloc][32 + hl * 16 + 8]);
        v8f st = (v8f){};
        st = wmma_bf16(aq[0], bk0, st);
        st = wmma_bf16(aq[1], bk1, st);
        s[t] = st;
      }

      // ---- causal mask ----
      float sv[2][8];
#pragma unroll
      for (int t = 0; t < 2; ++t) {
        int kk = kbase + t * 16 + ln16;
#pragma unroll
        for (int rr = 0; rr < 8; ++rr) {
          int qq = q0 + 8 * hl + rr;
          sv[t][rr] = (kk <= qq) ? s[t][rr] : -1e30f;
        }
      }

      // ---- online softmax update per row (DPP reductions) ----
#pragma unroll
      for (int rr = 0; rr < 8; ++rr) {
        float cm = red16_max(fmaxf(sv[0][rr], sv[1][rr]));
        float mn    = fmaxf(m_run[rr], cm);
        float alpha = __expf(m_run[rr] - mn);
        float p0 = __expf(sv[0][rr] - mn);
        float p1 = __expf(sv[1][rr] - mn);
        float rs = red16_sum(p0 + p1);
        l_run[rr] = l_run[rr] * alpha + rs;
        m_run[rr] = mn;
#pragma unroll
        for (int nt = 0; nt < 4; ++nt) o_acc[nt][rr] *= alpha;
        Ps[wv][8 * hl + rr][ln16]      = (bf16)p0;
        Ps[wv][8 * hl + rr][16 + ln16] = (bf16)p1;
      }

      // ---- P fragment (A layout) + O += P V ----
      v16bf pf = mk16(*(const v8bf*)&Ps[wv][ln16][hl * 16],
                      *(const v8bf*)&Ps[wv][ln16][hl * 16 + 8]);
#pragma unroll
      for (int nt = 0; nt < 4; ++nt) {
        v16bf vf = mk16(
            *(const v8bf*)&Vt[cur][nt * 16 + ln16][cch * 32 + hl * 16],
            *(const v8bf*)&Vt[cur][nt * 16 + ln16][cch * 32 + hl * 16 + 8]);
        o_acc[nt] = wmma_bf16(pf, vf, o_acc[nt]);
      }
    }

    wait_async_lds();      // next K/V block written
    __syncthreads();       // and everyone done with current block
    cur ^= 1;
  }

  // ---- normalize and write Y[b, q, h*D + d] (bf16) ----
#pragma unroll
  for (int nt = 0; nt < 4; ++nt) {
#pragma unroll
    for (int rr = 0; rr < 8; ++rr) {
      int q = q0 + 8 * hl + rr;
      int d = nt * 16 + ln16;
      Yb[((size_t)bIdx * T_ + q) * C_ + h * D_ + d] =
          (bf16)(o_acc[nt][rr] / l_run[rr]);
    }
  }
}

// ---------------------------------------------------------------------------
extern "C" void kernel_launch(void* const* d_in, const int* in_sizes, int n_in,
                              void* d_out, int out_size, void* d_ws,
                              size_t ws_size, hipStream_t stream) {
  const float* x      = (const float*)d_in[0];
  const float* W_attn = (const float*)d_in[1];
  const float* b_attn = (const float*)d_in[2];
  const float* W_proj = (const float*)d_in[3];
  const float* b_proj = (const float*)d_in[4];
  const float* bQ     = (const float*)d_in[5];
  const float* bK     = (const float*)d_in[6];
  const float* bV     = (const float*)d_in[7];
  float* out          = (float*)d_out;

  const size_t nX  = (size_t)B_ * T_ * C_;
  const size_t nWa = (size_t)C_ * 3 * C_;
  const size_t nWp = (size_t)C_ * C_;
  const size_t nQ  = (size_t)B_ * H_ * T_ * D_;

  char* ws = (char*)d_ws;
  bf16* Xb  = (bf16*)ws;            ws += nX  * sizeof(bf16);
  bf16* WaT = (bf16*)ws;            ws += nWa * sizeof(bf16);
  bf16* WpT = (bf16*)ws;            ws += nWp * sizeof(bf16);
  bf16* Qb  = (bf16*)ws;            ws += nQ  * sizeof(bf16);
  bf16* Kb  = (bf16*)ws;            ws += nQ  * sizeof(bf16);
  bf16* Vt  = (bf16*)ws;            ws += nQ  * sizeof(bf16);
  bf16* Yb  = (bf16*)ws;            ws += nX  * sizeof(bf16);

  const int M = B_ * T_;   // 8192

  convert_inputs<<<2048, 256, 0, stream>>>(x, W_attn, W_proj, Xb, WaT, WpT);

  // QKV: [8192 x 1024] @ [1024 x 3072]
  gemm_wmma<0><<<dim3(3 * C_ / BN, M / BM), 256, 0, stream>>>(
      Xb, WaT, b_attn, bQ, bK, bV, Qb, Kb, Vt, nullptr, M, 3 * C_, C_);

  // Flash attention over (16 q-blocks, B*H)
  flash_attn<<<dim3(T_ / 128, B_ * H_), 256, 0, stream>>>(Qb, Kb, Vt, Yb);

  // Projection: [8192 x 1024] @ [1024 x 1024] + b_proj -> fp32 out
  gemm_wmma<1><<<dim3(C_ / BN, M / BM), 256, 0, stream>>>(
      Yb, WpT, b_proj, nullptr, nullptr, nullptr,
      nullptr, nullptr, nullptr, out, M, C_, C_);
}
